// BeforeAfterPoolBlock_37830071943289
// MI455X (gfx1250) — matrix operand order repaired and checked by reading, not compiled, not run on hardware
//
#include <hip/hip_runtime.h>
#include <hip/hip_bf16.h>

typedef __attribute__((ext_vector_type(16))) __bf16 v16bf;
typedef __attribute__((ext_vector_type(8)))  __bf16 v8bf;
typedef __attribute__((ext_vector_type(8)))  float  v8f;

#define N_ROWS 8192
#define DIM    1024
#define DFF    4096
#define K3     3072      // 3*DIM
#define SEG    512
#define NSEG   16        // 8192/512

// ---------- CDNA5 async global->LDS copy (ASYNCcnt-tracked, bypasses VGPRs) ----------
__device__ __forceinline__ void async_cp16(void* lds, const void* gptr) {
    // low 32 bits of a flat LDS address are the LDS byte offset
    unsigned lofs = (unsigned)(unsigned long long)lds;
    unsigned long long ga = (unsigned long long)gptr;
    asm volatile("global_load_async_to_lds_b128 %0, %1, off"
                 :: "v"(lofs), "v"(ga) : "memory");
}
__device__ __forceinline__ void wait_async_le4() {
    asm volatile("s_wait_asynccnt 0x4" ::: "memory");
}
__device__ __forceinline__ void wait_async_0() {
    asm volatile("s_wait_asynccnt 0x0" ::: "memory");
}

// ---------------- f32 -> bf16 convert (grid-stride) ----------------
__global__ void f32_to_bf16_kernel(const float* __restrict__ in,
                                   __bf16* __restrict__ out, int n) {
    int i = blockIdx.x * blockDim.x + threadIdx.x;
    int stride = gridDim.x * blockDim.x;
    for (; i < n; i += stride) out[i] = (__bf16)in[i];
}

// ---------------- segment max per column ----------------
__global__ void segmax_kernel(const float* __restrict__ x,
                              float* __restrict__ segmax) {
    const int c = blockIdx.x * 64 + threadIdx.x;   // column
    const int s = blockIdx.y;                      // segment
    const float* p = x + (size_t)s * SEG * DIM + c;
    float m = -__builtin_inff();
    for (int i = 0; i < SEG; ++i) m = fmaxf(m, p[(size_t)i * DIM]);
    segmax[s * DIM + c] = m;
}

// ---------------- exclusive prefix/suffix over segment maxes ----------------
__global__ void segscan_kernel(const float* __restrict__ segmax,
                               float* __restrict__ pre,
                               float* __restrict__ suf) {
    const int c = blockIdx.x * 256 + threadIdx.x;
    float run = -__builtin_inff();
    for (int s = 0; s < NSEG; ++s) {
        pre[s * DIM + c] = run;
        run = fmaxf(run, segmax[s * DIM + c]);
    }
    run = -__builtin_inff();
    for (int s = NSEG - 1; s >= 0; --s) {
        suf[s * DIM + c] = run;
        run = fmaxf(run, segmax[s * DIM + c]);
    }
}

// ------------- build concatenated [x | before | after] in bf16 -------------
__global__ void build_a3_kernel(const float* __restrict__ x,
                                const float* __restrict__ pre,
                                const float* __restrict__ suf,
                                __bf16* __restrict__ a3) {
    const int c = blockIdx.x * 64 + threadIdx.x;
    const int s = blockIdx.y;
    const int base = s * SEG;

    float carry = pre[s * DIM + c];
    float run = -__builtin_inff();
    for (int i = 0; i < SEG; ++i) {
        const int gi = base + i;
        const float bef = (gi == 0) ? 0.0f : fmaxf(carry, run);
        const float v = x[(size_t)gi * DIM + c];
        run = fmaxf(run, v);
        a3[(size_t)gi * K3 + c]       = (__bf16)v;
        a3[(size_t)gi * K3 + DIM + c] = (__bf16)bef;
    }
    carry = suf[s * DIM + c];
    run = -__builtin_inff();
    for (int i = SEG - 1; i >= 0; --i) {
        const int gi = base + i;
        const float aft = (gi == N_ROWS - 1) ? 0.0f : fmaxf(carry, run);
        const float v = x[(size_t)gi * DIM + c];
        run = fmaxf(run, v);
        a3[(size_t)gi * K3 + 2 * DIM + c] = (__bf16)aft;
    }
}

// ---------------- tiled WMMA bf16 GEMM: C = A[M,K] @ B[N,K]^T ----------------
// Block: 256 threads = 8 waves. Block tile 128x128, BK=32, TRIPLE-buffered LDS
// filled by global_load_async_to_lds_b128 (ASYNCcnt), prefetch distance 2,
// ONE barrier per K-step (issue-after-barrier makes the WAR race impossible).
// Waves in 4(M) x 2(N) grid; each wave: 2 A-frags x 4 B-frags = 8 WMMAs/K-step.
// EPI==0: out_bf = relu(acc + bias)            (GEMM1 -> h, bf16)
// EPI==1: out_f  = (acc + bias)*gamma + resid  (GEMM2 -> final, f32)
template <int EPI>
__global__ __launch_bounds__(256) void gemm_bf16_kernel(
    const __bf16* __restrict__ A, const __bf16* __restrict__ B,
    const float* __restrict__ bias,
    __bf16* __restrict__ outb, float* __restrict__ outf,
    const float* __restrict__ gamma, const float* __restrict__ resid,
    int M, int N, int K) {
    constexpr int BM = 128, BN = 128, BK = 32;
    constexpr int LDT = 40;                 // padded LDS row (bf16), 80B = 5*16B
    __shared__ __bf16 As[3][BM * LDT];
    __shared__ __bf16 Bs[3][BM * LDT];

    const int t    = threadIdx.x;
    const int lane = t & 31;
    const int wave = t >> 5;
    const int m0 = blockIdx.y * BM;
    const int n0 = blockIdx.x * BN;
    const int wm = wave & 3;                // 0..3 -> 32-row strip
    const int wn = wave >> 2;               // 0..1 -> 64-col strip

    // cooperative async loads: 128 rows x 32 cols per tile, 2 chunks/thread/tile
    const int r0 = t >> 2;                  // 0..63
    const int c0 = (t & 3) * 8;             // 0,8,16,24

    // fragment addressing (16-bit WMMA layouts, wave32)
    const int alane = lane & 15, abase = (lane >> 4) * 16;
    const int bcol  = lane & 15, bbase = (lane >> 4) * 16;

    v8f acc[2][4] = {};

    const int KSTEPS = K / BK;

    auto issueTile = [&](int buf, int k0) {
#pragma unroll
        for (int i = 0; i < 2; ++i) {
            const int rr = r0 + 64 * i;
            async_cp16(&As[buf][rr * LDT + c0],
                       &A[(size_t)(m0 + rr) * K + k0 + c0]);
            async_cp16(&Bs[buf][rr * LDT + c0],
                       &B[(size_t)(n0 + rr) * K + k0 + c0]);
        }
    };

    auto computeStep = [&](int buf) {
        v16bf a[2], b[4];
#pragma unroll
        for (int fi = 0; fi < 2; ++fi) {
            const __bf16* ap = &As[buf][(wm * 32 + fi * 16 + alane) * LDT + abase];
            v8bf lo = *(const v8bf*)ap;
            v8bf hi = *(const v8bf*)(ap + 8);
            a[fi] = __builtin_shufflevector(lo, hi,
                        0,1,2,3,4,5,6,7,8,9,10,11,12,13,14,15);
        }
#pragma unroll
        for (int fj = 0; fj < 4; ++fj)
            b[fj] = *(const v16bf*)&Bs[buf][(wn * 64 + fj * 16 + bcol) * LDT + bbase];
#pragma unroll
        for (int fi = 0; fi < 2; ++fi)
#pragma unroll
            for (int fj = 0; fj < 4; ++fj)
                acc[fi][fj] = __builtin_amdgcn_wmma_f32_16x16x32_bf16(
                    false, a[fi], false, b[fj], (short)0, acc[fi][fj],
                    false, false);
    };

    // prologue: prefetch tiles 0 and 1 (KSTEPS >= 2 always here)
    issueTile(0, 0);
    issueTile(1, BK);

    int next_k = 2 * BK;
    int bi = 2;   // buffer to issue next
    int bc = 0;   // buffer to compute from
    for (int ks = 0; ks < KSTEPS - 1; ++ks) {
        wait_async_le4();            // oldest 4 async ops (= tile bc) landed
        __syncthreads();             // all waves' tile-bc data visible; iter ks-1 reads done
        if (next_k < K) {
            issueTile(bi, next_k);   // DMA overlaps the WMMAs below
            next_k += BK;
            bi = (bi == 2) ? 0 : bi + 1;
        }
        computeStep(bc);
        bc = (bc == 2) ? 0 : bc + 1;
    }
    wait_async_0();
    __syncthreads();
    computeStep(bc);

    // epilogue: C/D layout -> VGPR r: lanes 0-15 = M r, lanes 16-31 = M r+8
    const int cl = lane & 15;
    const int rsel = (lane >> 4) ? 8 : 0;
#pragma unroll
    for (int fi = 0; fi < 2; ++fi) {
        const int rbase = m0 + wm * 32 + fi * 16 + rsel;
#pragma unroll
        for (int fj = 0; fj < 4; ++fj) {
            const int col = n0 + wn * 64 + fj * 16 + cl;
#pragma unroll
            for (int r = 0; r < 8; ++r) {
                const int row = rbase + r;
                float v = acc[fi][fj][r] + bias[col];
                if (EPI == 0) {
                    outb[(size_t)row * N + col] = (__bf16)fmaxf(v, 0.0f);
                } else {
                    outf[(size_t)row * N + col] =
                        v * gamma[col] + resid[(size_t)row * N + col];
                }
            }
        }
    }
}

// ---------------- per-row LayerNorm over DFF, in place on bf16 h ----------------
__global__ __launch_bounds__(256) void ln_kernel(__bf16* __restrict__ h,
                                                 const float* __restrict__ w,
                                                 const float* __restrict__ b) {
    __shared__ float red[256];
    const int row = blockIdx.x;
    const int t = threadIdx.x;
    __bf16* hr = h + (size_t)row * DFF;

    float vals[16];
    float s = 0.f, s2 = 0.f;
#pragma unroll
    for (int i = 0; i < 16; ++i) {
        float v = (float)hr[t * 16 + i];
        vals[i] = v;
        s += v;
        s2 += v * v;
    }
    red[t] = s;
    __syncthreads();
    for (int o = 128; o > 0; o >>= 1) {
        if (t < o) red[t] += red[t + o];
        __syncthreads();
    }
    const float mu = red[0] * (1.0f / DFF);
    __syncthreads();
    red[t] = s2;
    __syncthreads();
    for (int o = 128; o > 0; o >>= 1) {
        if (t < o) red[t] += red[t + o];
        __syncthreads();
    }
    const float var = red[0] * (1.0f / DFF) - mu * mu;
    const float rs = rsqrtf(var + 1e-6f);
#pragma unroll
    for (int i = 0; i < 16; ++i) {
        const int c = t * 16 + i;
        hr[c] = (__bf16)((vals[i] - mu) * rs * w[c] + b[c]);
    }
}

extern "C" void kernel_launch(void* const* d_in, const int* in_sizes, int n_in,
                              void* d_out, int out_size, void* d_ws, size_t ws_size,
                              hipStream_t stream) {
    const float* x     = (const float*)d_in[0];   // [8192,1024]
    const float* W1    = (const float*)d_in[1];   // [4096,3072]
    const float* b1    = (const float*)d_in[2];   // [4096]
    const float* ln_w  = (const float*)d_in[3];   // [4096]
    const float* ln_b  = (const float*)d_in[4];   // [4096]
    const float* W2    = (const float*)d_in[5];   // [1024,4096]
    const float* b2    = (const float*)d_in[6];   // [1024]
    const float* gamma = (const float*)d_in[7];   // [1024]
    float* out = (float*)d_out;                   // [8192,1024]

    char* w = (char*)d_ws;
    __bf16* A3  = (__bf16*)w;  w += (size_t)N_ROWS * K3 * 2;       // 48 MB
    __bf16* W1b = (__bf16*)w;  w += (size_t)DFF * K3 * 2;          // 24 MB
    __bf16* W2b = (__bf16*)w;  w += (size_t)DIM * DFF * 2;         //  8 MB
    __bf16* H   = (__bf16*)w;  w += (size_t)N_ROWS * DFF * 2;      // 64 MB
    float* segmax = (float*)w; w += NSEG * DIM * 4;
    float* segpre = (float*)w; w += NSEG * DIM * 4;
    float* segsuf = (float*)w; w += NSEG * DIM * 4;

    // weights -> bf16
    f32_to_bf16_kernel<<<2048, 256, 0, stream>>>(W1, W1b, DFF * K3);
    f32_to_bf16_kernel<<<2048, 256, 0, stream>>>(W2, W2b, DIM * DFF);

    // prefix/suffix max scan + bf16 [x|before|after]
    segmax_kernel<<<dim3(DIM / 64, NSEG), 64, 0, stream>>>(x, segmax);
    segscan_kernel<<<DIM / 256, 256, 0, stream>>>(segmax, segpre, segsuf);
    build_a3_kernel<<<dim3(DIM / 64, NSEG), 64, 0, stream>>>(x, segpre, segsuf, A3);

    // GEMM1 + bias + ReLU -> H (bf16)
    gemm_bf16_kernel<0><<<dim3(DFF / 128, N_ROWS / 128), 256, 0, stream>>>(
        A3, W1b, b1, H, nullptr, nullptr, nullptr, N_ROWS, DFF, K3);

    // LayerNorm in place on H
    ln_kernel<<<N_ROWS, 256, 0, stream>>>(H, ln_w, ln_b);

    // GEMM2 + bias, *gamma, +residual -> out (f32)
    gemm_bf16_kernel<1><<<dim3(DIM / 128, N_ROWS / 128), 256, 0, stream>>>(
        H, W2b, b2, nullptr, out, gamma, x, N_ROWS, DIM, DFF);
}
